// GeminiFlashAttention_13683765805443
// MI455X (gfx1250) — compile-verified, matrix operand order checked by reading, not run on hardware
//
#include <hip/hip_runtime.h>
#include <hip/hip_bf16.h>

typedef __attribute__((ext_vector_type(16))) _Float16 v16h;
typedef __attribute__((ext_vector_type(8)))  float    v8f;

#define NB      2
#define SEQ     2048
#define DIMM    1024
#define NHEADS  16
#define HDIM    64
#define ROWS    (NB * SEQ)          // 4096

#define WMMA16(a, b, c) __builtin_amdgcn_wmma_f32_16x16x32_f16( \
    false, (a), false, (b), (short)0, (c), false, false)

// ---------------------------------------------------------------------------
// Fragment gather. 16-bit A-matrix 16x32 layout (ISA 7.12.2):
//   lanes 0-15 : row M = lane,    K = {0..7, 16..23}
//   lanes 16-31: row M = lane-16, K = {8..15, 24..31}
// B fragments are the same layout applied to B^T (column-major per lane),
// so the same loader serves both when the matrix is stored "N-major".
// Pairs are contiguous; backend coalesces each fragment into b128 loads.
// ---------------------------------------------------------------------------
__device__ __forceinline__ v16h load_frag(const _Float16* p, int row0, int kBase,
                                          int ld, int lane) {
    const _Float16* rp = p + (size_t)(row0 + (lane & 15)) * ld + kBase
                           + ((lane & 16) ? 8 : 0);
    v16h a;
#pragma unroll
    for (int t = 0; t < 8; ++t) {
        int kk = (t < 4) ? (2 * t) : (2 * t + 8);   // 0,2,4,6,16,18,20,22
        a[2 * t]     = rp[kk];
        a[2 * t + 1] = rp[kk + 1];
    }
    return a;
}

// ---------------------------------------------------------------------------
// Elementwise fp32 -> fp16 convert
// ---------------------------------------------------------------------------
__global__ void k_cvt_f16(const float* __restrict__ in, _Float16* __restrict__ out, int n) {
    int i = blockIdx.x * blockDim.x + threadIdx.x;
    if (i < n) out[i] = (_Float16)in[i];
}

// in [K,N] fp32 -> out [N,K] fp16 (transpose+convert so GEMM B gathers are contiguous)
__global__ void k_cvt_transpose_f16(const float* __restrict__ in, _Float16* __restrict__ out,
                                    int K, int N) {
    int i = blockIdx.x * blockDim.x + threadIdx.x;
    if (i < K * N) {
        int nn = i / K, kk = i - nn * K;
        out[i] = (_Float16)in[(size_t)kk * N + nn];
    }
}

// ---------------------------------------------------------------------------
// QKV GEMM: C[4096,3072] = x16[4096,1024] @ Wqkv, Wqkv given transposed [3072,1024].
// Per wave: 32 rows x 64 cols (2 A-frags, 4 B-frags, 8 WMMA per k-step).
// Block = 256 thr (8 waves) covers 256 rows x 64 cols.
// Epilogue adds bias, scatters into Q [b,h,s,hd], K [b,h,s,hd], Vt [b,h,hd,s] (f16).
// ---------------------------------------------------------------------------
__global__ __launch_bounds__(256) void k_gemm_qkv(
        const _Float16* __restrict__ x16, const _Float16* __restrict__ wT,
        const float* __restrict__ bias,
        _Float16* __restrict__ Q, _Float16* __restrict__ K, _Float16* __restrict__ Vt) {
    int lane = threadIdx.x & 31;
    int wave = threadIdx.x >> 5;
    int row0 = (blockIdx.y * 8 + wave) * 32;
    int col0 = blockIdx.x * 64;

    v8f acc[2][4] = {{v8f{}, v8f{}, v8f{}, v8f{}}, {v8f{}, v8f{}, v8f{}, v8f{}}};
    for (int k0 = 0; k0 < DIMM; k0 += 32) {
        v16h a0 = load_frag(x16, row0,      k0, DIMM, lane);
        v16h a1 = load_frag(x16, row0 + 16, k0, DIMM, lane);
#pragma unroll
        for (int n = 0; n < 4; ++n) {
            v16h b = load_frag(wT, col0 + n * 16, k0, DIMM, lane);
            acc[0][n] = WMMA16(a0, b, acc[0][n]);
            acc[1][n] = WMMA16(a1, b, acc[1][n]);
        }
    }
#pragma unroll
    for (int r = 0; r < 2; ++r) {
#pragma unroll
        for (int n = 0; n < 4; ++n) {
            int col   = col0 + n * 16 + (lane & 15);
            int which = col >> 10;           // 0=Q 1=K 2=V
            int d     = col & 1023;
            int h     = d >> 6;
            int hd    = d & 63;
            float bv  = bias[col];
#pragma unroll
            for (int j = 0; j < 8; ++j) {
                int row = row0 + r * 16 + j + ((lane & 16) ? 8 : 0);
                int bb  = row >> 11;
                int s   = row & 2047;
                _Float16 hv = (_Float16)(acc[r][n][j] + bv);
                size_t bhs = ((size_t)bb * NHEADS + h);
                if (which == 0)      Q [(bhs * SEQ + s) * HDIM + hd] = hv;
                else if (which == 1) K [(bhs * SEQ + s) * HDIM + hd] = hv;
                else                 Vt[(bhs * HDIM + hd) * SEQ + s] = hv;
            }
        }
    }
}

// ---------------------------------------------------------------------------
// Flash attention. grid = (16 q-blocks of 128, 32 b*h). Block = 256 thr (8 waves),
// each wave owns 16 query rows, streams keys in chunks of 64:
//   8 WMMA for S = Q*K^T (4 key tiles x 2 hd k-steps),
//   streaming softmax (shfl reductions within 16-lane halves, 1 rescale / 64 keys),
//   P (16x64) staged via LDS to convert C-layout -> A-layout (2 A frags),
//   8 WMMA for O += P*V.
// ---------------------------------------------------------------------------
__global__ __launch_bounds__(256) void k_flash_attn(
        const _Float16* __restrict__ Qg, const _Float16* __restrict__ Kg,
        const _Float16* __restrict__ Vtg, _Float16* __restrict__ attn16) {
    int lane = threadIdx.x & 31;
    int wave = threadIdx.x >> 5;
    int bh   = blockIdx.y;                               // b*16 + h
    const _Float16* Qp  = Qg  + (size_t)bh * SEQ * HDIM;
    const _Float16* Kp  = Kg  + (size_t)bh * SEQ * HDIM;
    const _Float16* Vtp = Vtg + (size_t)bh * HDIM * SEQ;
    int q0 = (blockIdx.x * 8 + wave) * 16;

    v16h qa0 = load_frag(Qp, q0, 0,  HDIM, lane);
    v16h qa1 = load_frag(Qp, q0, 32, HDIM, lane);

    float m[8], l[8];
#pragma unroll
    for (int j = 0; j < 8; ++j) { m[j] = -1.0e30f; l[j] = 0.0f; }
    v8f o[4] = {v8f{}, v8f{}, v8f{}, v8f{}};

    __shared__ _Float16 Plds[8][16 * 64];
    _Float16* myP = Plds[wave];
    const float scale = 0.125f;                          // 1/sqrt(64)

    for (int kc = 0; kc < SEQ; kc += 64) {
        v8f s[4] = {v8f{}, v8f{}, v8f{}, v8f{}};
#pragma unroll
        for (int t = 0; t < 4; ++t) {
            v16h b0 = load_frag(Kp, kc + t * 16, 0,  HDIM, lane);
            s[t] = WMMA16(qa0, b0, s[t]);
            v16h b1 = load_frag(Kp, kc + t * 16, 32, HDIM, lane);
            s[t] = WMMA16(qa1, b1, s[t]);
        }
#pragma unroll
        for (int j = 0; j < 8; ++j) {
            float v0 = s[0][j] * scale;
            float v1 = s[1][j] * scale;
            float v2 = s[2][j] * scale;
            float v3 = s[3][j] * scale;
            float mx = fmaxf(fmaxf(v0, v1), fmaxf(v2, v3));
#pragma unroll
            for (int msk = 8; msk >= 1; msk >>= 1)
                mx = fmaxf(mx, __shfl_xor(mx, msk, 32));
            float mnew = fmaxf(m[j], mx);
            float p0 = __expf(v0 - mnew);
            float p1 = __expf(v1 - mnew);
            float p2 = __expf(v2 - mnew);
            float p3 = __expf(v3 - mnew);
            float ps = (p0 + p1) + (p2 + p3);
#pragma unroll
            for (int msk = 8; msk >= 1; msk >>= 1)
                ps += __shfl_xor(ps, msk, 32);
            float alpha = __expf(m[j] - mnew);
            l[j] = l[j] * alpha + ps;
            m[j] = mnew;
#pragma unroll
            for (int n = 0; n < 4; ++n) o[n][j] *= alpha;
            int base = (j + ((lane & 16) ? 8 : 0)) * 64 + (lane & 15);
            myP[base]      = (_Float16)p0;
            myP[base + 16] = (_Float16)p1;
            myP[base + 32] = (_Float16)p2;
            myP[base + 48] = (_Float16)p3;
        }
        asm volatile("s_wait_dscnt 0" ::: "memory");     // ds stores visible before re-gather
        v16h pa0 = load_frag(myP, 0, 0,  64, lane);
        v16h pa1 = load_frag(myP, 0, 32, 64, lane);
#pragma unroll
        for (int n = 0; n < 4; ++n) {
            v16h vb0 = load_frag(Vtp, n * 16, kc,      SEQ, lane);
            o[n] = WMMA16(pa0, vb0, o[n]);
            v16h vb1 = load_frag(Vtp, n * 16, kc + 32, SEQ, lane);
            o[n] = WMMA16(pa1, vb1, o[n]);
        }
        asm volatile("" ::: "memory");                   // keep next chunk's LDS stores behind loads
    }

    // write attn output as f16 [b, s, h*64+hd] for the projection GEMM
    int b = bh >> 4, h = bh & 15;
#pragma unroll
    for (int n = 0; n < 4; ++n) {
        int col = h * HDIM + n * 16 + (lane & 15);
#pragma unroll
        for (int j = 0; j < 8; ++j) {
            int sr = q0 + j + ((lane & 16) ? 8 : 0);
            attn16[((size_t)b * SEQ + sr) * DIMM + col] = (_Float16)(o[n][j] / l[j]);
        }
    }
}

// ---------------------------------------------------------------------------
// Output projection: out[4096,1024] = attn16 @ Wproj (given transposed) + bias, fp32 out.
// Same 32x64-per-wave tiling as the QKV GEMM.
// ---------------------------------------------------------------------------
__global__ __launch_bounds__(256) void k_gemm_proj(
        const _Float16* __restrict__ A, const _Float16* __restrict__ wT,
        const float* __restrict__ bias, float* __restrict__ out) {
    int lane = threadIdx.x & 31;
    int wave = threadIdx.x >> 5;
    int row0 = (blockIdx.y * 8 + wave) * 32;
    int col0 = blockIdx.x * 64;

    v8f acc[2][4] = {{v8f{}, v8f{}, v8f{}, v8f{}}, {v8f{}, v8f{}, v8f{}, v8f{}}};
    for (int k0 = 0; k0 < DIMM; k0 += 32) {
        v16h a0 = load_frag(A, row0,      k0, DIMM, lane);
        v16h a1 = load_frag(A, row0 + 16, k0, DIMM, lane);
#pragma unroll
        for (int n = 0; n < 4; ++n) {
            v16h b = load_frag(wT, col0 + n * 16, k0, DIMM, lane);
            acc[0][n] = WMMA16(a0, b, acc[0][n]);
            acc[1][n] = WMMA16(a1, b, acc[1][n]);
        }
    }
#pragma unroll
    for (int r = 0; r < 2; ++r) {
#pragma unroll
        for (int n = 0; n < 4; ++n) {
            int col  = col0 + n * 16 + (lane & 15);
            float bv = bias[col];
#pragma unroll
            for (int j = 0; j < 8; ++j) {
                int row = row0 + r * 16 + j + ((lane & 16) ? 8 : 0);
                out[(size_t)row * DIMM + col] = acc[r][n][j] + bv;
            }
        }
    }
}

// ---------------------------------------------------------------------------
extern "C" void kernel_launch(void* const* d_in, const int* in_sizes, int n_in,
                              void* d_out, int out_size, void* d_ws, size_t ws_size,
                              hipStream_t stream) {
    (void)in_sizes; (void)n_in; (void)out_size; (void)ws_size;
    const float* x      = (const float*)d_in[0];
    const float* W_qkv  = (const float*)d_in[1];
    const float* b_qkv  = (const float*)d_in[2];
    const float* W_proj = (const float*)d_in[3];
    const float* b_proj = (const float*)d_in[4];
    float* out = (float*)d_out;

    char* ws = (char*)d_ws;
    _Float16* x16     = (_Float16*)ws;  ws += (size_t)ROWS * DIMM * 2;           //  8 MB
    _Float16* wqkvT   = (_Float16*)ws;  ws += (size_t)3 * DIMM * DIMM * 2;       //  6 MB
    _Float16* wprojT  = (_Float16*)ws;  ws += (size_t)DIMM * DIMM * 2;           //  2 MB
    _Float16* Qf      = (_Float16*)ws;  ws += (size_t)ROWS * DIMM * 2;           //  8 MB
    _Float16* Kf      = (_Float16*)ws;  ws += (size_t)ROWS * DIMM * 2;           //  8 MB
    _Float16* Vt      = (_Float16*)ws;  ws += (size_t)ROWS * DIMM * 2;           //  8 MB
    _Float16* attn16  = (_Float16*)ws;  ws += (size_t)ROWS * DIMM * 2;           //  8 MB

    // 1) precision convert + weight transposes
    {
        int n = ROWS * DIMM;
        k_cvt_f16<<<(n + 255) / 256, 256, 0, stream>>>(x, x16, n);
    }
    {
        int n = 3 * DIMM * DIMM;  // Wqkv [1024,3072] -> [3072,1024]
        k_cvt_transpose_f16<<<(n + 255) / 256, 256, 0, stream>>>(W_qkv, wqkvT, DIMM, 3 * DIMM);
    }
    {
        int n = DIMM * DIMM;      // Wproj [1024,1024] -> [1024,1024]^T
        k_cvt_transpose_f16<<<(n + 255) / 256, 256, 0, stream>>>(W_proj, wprojT, DIMM, DIMM);
    }

    // 2) QKV projection with bias -> Q, K, V^T (f16, attention-friendly layouts)
    k_gemm_qkv<<<dim3(48, 16), 256, 0, stream>>>(x16, wqkvT, b_qkv, Qf, Kf, Vt);

    // 3) flash attention -> attn16 [b, s, dim]
    k_flash_attn<<<dim3(16, 32), 256, 0, stream>>>(Qf, Kf, Vt, attn16);

    // 4) output projection + bias -> fp32 out
    k_gemm_proj<<<dim3(16, 16), 256, 0, stream>>>(attn16, wprojT, b_proj, out);
}
